// GGNN_13357348291347
// MI455X (gfx1250) — compile-verified, matrix operand order checked by reading, not run on hardware
//
#include <hip/hip_runtime.h>
#include <hip/hip_bf16.h>
#include <math.h>

#define HD 512
#define NE 4096
#define NP 65536
#define EPSV 1e-4f

typedef _Float16 h8  __attribute__((ext_vector_type(8)));
typedef _Float16 h16 __attribute__((ext_vector_type(16)));
typedef float    f8  __attribute__((ext_vector_type(8)));

// ---------------- elementwise / graph kernels ----------------

__global__ void k_cvt(const float* __restrict__ s, _Float16* __restrict__ d, long n) {
    long i = (long)blockIdx.x * blockDim.x + threadIdx.x;
    if (i < n) d[i] = (_Float16)s[i];
}

__global__ void k_cvt_dual(const float* __restrict__ s, float* __restrict__ d32,
                           _Float16* __restrict__ d16, long n) {
    long i = (long)blockIdx.x * blockDim.x + threadIdx.x;
    if (i < n) { float v = s[i]; d32[i] = v; d16[i] = (_Float16)v; }
}

__global__ void k_zero(float* __restrict__ p, long n) {
    long i = (long)blockIdx.x * blockDim.x + threadIdx.x;
    if (i < n) p[i] = 0.f;
}

__global__ void k_counts(const int* __restrict__ rel, float* __restrict__ cs,
                         float* __restrict__ co) {
    int p = blockIdx.x * blockDim.x + threadIdx.x;
    if (p < NP) {
        atomicAdd(cs + rel[2 * p], 1.f);
        atomicAdd(co + rel[2 * p + 1], 1.f);
    }
}

__global__ void k_segsum(const int* __restrict__ rel, const float* __restrict__ ms,
                         float* __restrict__ ss, float* __restrict__ so) {
    long i = (long)blockIdx.x * blockDim.x + threadIdx.x;
    if (i < (long)NP * 128) {
        int p = (int)(i >> 7), c = (int)(i & 127);
        float v = ms[i];
        atomicAdd(ss + (long)rel[2 * p] * 128 + c, v);
        atomicAdd(so + (long)rel[2 * p + 1] * 128 + c, v);
    }
}

// inc_ent[e, w*128+c] = normalize( seg_{w}[e,:]/max(cnt,1) )
__global__ void k_inc_ent(const float* __restrict__ ss, const float* __restrict__ so,
                          const float* __restrict__ cs, const float* __restrict__ co,
                          _Float16* __restrict__ inc) {
    int i = blockIdx.x * blockDim.x + threadIdx.x;
    if (i >= NE * 2) return;
    int e = i >> 1, w = i & 1;
    const float* src = (w ? so : ss) + (long)e * 128;
    float inv = 1.f / fmaxf((w ? co : cs)[e], 1.f);
    float sq = 0.f;
    for (int c = 0; c < 128; c++) { float v = src[c] * inv; sq += v * v; }
    float sc = inv * rsqrtf(fmaxf(sq, EPSV));
    for (int c = 0; c < 128; c++)
        inc[(long)e * 256 + w * 128 + c] = (_Float16)(src[c] * sc);
}

// inc_pred[p, w*128+c] = normalize( ms_ent[rel[p,w], :] )
__global__ void k_inc_pred(const int* __restrict__ rel, const float* __restrict__ msE,
                           _Float16* __restrict__ inc) {
    int i = blockIdx.x * blockDim.x + threadIdx.x;
    if (i >= NP * 2) return;
    int p = i >> 1, w = i & 1;
    const float* src = msE + (long)rel[2 * p + w] * 128;
    float sq = 0.f;
    for (int c = 0; c < 128; c++) { float v = src[c]; sq += v * v; }
    float sc = rsqrtf(fmaxf(sq, EPSV));
    for (int c = 0; c < 128; c++)
        inc[(long)p * 256 + w * 128 + c] = (_Float16)(src[c] * sc);
}

// rn = r * n  (f16, feeds the 5u GEMM)
__global__ void k_rn(const float* __restrict__ r, const float* __restrict__ n,
                     _Float16* __restrict__ rn, long cnt) {
    long i = (long)blockIdx.x * blockDim.x + threadIdx.x;
    if (i < cnt) rn[i] = (_Float16)(r[i] * n[i]);
}

// nodes = (1-z)*n + z*h ; write f32 + f16
__global__ void k_combine(const float* __restrict__ z, const float* __restrict__ h,
                          float* __restrict__ n32, _Float16* __restrict__ n16, long cnt) {
    long i = (long)blockIdx.x * blockDim.x + threadIdx.x;
    if (i < cnt) {
        float zi = z[i];
        float v = (1.f - zi) * n32[i] + zi * h[i];
        n32[i] = v;
        n16[i] = (_Float16)v;
    }
}

// ---------------- WMMA GEMM ----------------

__device__ __forceinline__ h16 cat8(h8 lo, h8 hi) {
    h16 r;
#pragma unroll
    for (int i = 0; i < 8; i++) { r[i] = lo[i]; r[i + 8] = hi[i]; }
    return r;
}

__device__ __forceinline__ float actf(float v, int act) {
    if (act == 1) return v > 0.f ? v : 0.f;
    if (act == 2) return 1.f / (1.f + expf(-v));
    if (act == 3) return tanhf(v);
    return v;
}

// C[M,N] = act( A[M,K](f16) @ B[K,N](f16) + bias + Cadd ), outputs f32/f16.
// Block tile 128x128, BK=32; 8 wave32s in a 4(M) x 2(N) grid; each wave owns a
// 32x64 patch = 2x4 v_wmma_f32_16x16x32_f16 tiles.
// FULL=true: M%128==0 and N%128==0 -> no masks, vector global loads.
template <bool FULL>
__launch_bounds__(256)
__global__ void k_gemm(const _Float16* __restrict__ A, const _Float16* __restrict__ B,
                       const float* __restrict__ bias, const float* __restrict__ Cadd,
                       float* __restrict__ outF, _Float16* __restrict__ outH,
                       int M, int N, int K, int act) {
    __shared__ __align__(16) _Float16 As[128][40]; // [m][k], row = 80B (16B multiple)
    __shared__ __align__(16) _Float16 Bs[128][40]; // transposed: [n][k]

    const int tid  = threadIdx.x;
    const int lane = tid & 31;
    const int wave = tid >> 5;
    const int m0 = blockIdx.y * 128;
    const int n0 = blockIdx.x * 128;
    const int wm = (wave >> 1) * 32; // 4 waves along M
    const int wn = (wave & 1) * 64;  // 2 waves along N
    const int lm  = lane & 15;
    const int hi  = lane >> 4;
    const int kb0 = hi * 8;       // K-halves [kb0, kb0+8)
    const int kb1 = 16 + hi * 8;  // K-halves [kb1, kb1+8)

    f8 acc[2][4];
#pragma unroll
    for (int i = 0; i < 2; i++)
#pragma unroll
        for (int j = 0; j < 4; j++)
#pragma unroll
            for (int v = 0; v < 8; v++) acc[i][j][v] = 0.f;

    const int arow = tid >> 1, achk = tid & 1;  // A: 128 rows x 2 chunks of 16 halves
    const int brow = tid >> 3, bchk = tid & 7;  // B: 32 k-rows x 8 chunks of 16 halves

    for (int k0 = 0; k0 < K; k0 += 32) {
        __syncthreads();
        { // A tile fill (unmasked in FULL path: M is a multiple of 128)
            const _Float16* src = A + (long)(m0 + arow) * K + k0 + achk * 16;
            h8 v0 = *(const h8*)src;
            h8 v1 = *(const h8*)(src + 8);
            if (k0 + 32 < K) __builtin_prefetch((const void*)(src + 32), 0, 0);
            *(h8*)&As[arow][achk * 16]     = v0;
            *(h8*)&As[arow][achk * 16 + 8] = v1;
        }
        { // B tile fill, transposed into [n][k]
            int gk = k0 + brow;
            int gn = n0 + bchk * 16;
            const _Float16* src = B + (long)gk * N + gn;
            _Float16 t[16];
            if (FULL) {
                h8 v0 = *(const h8*)src;
                h8 v1 = *(const h8*)(src + 8);
#pragma unroll
                for (int j = 0; j < 8; j++) { t[j] = v0[j]; t[j + 8] = v1[j]; }
            } else {
#pragma unroll
                for (int j = 0; j < 16; j++) t[j] = (gn + j < N) ? src[j] : (_Float16)0.f;
            }
#pragma unroll
            for (int j = 0; j < 16; j++) Bs[bchk * 16 + j][brow] = t[j];
        }
        __syncthreads();

        h16 af[2], bf[4];
#pragma unroll
        for (int i = 0; i < 2; i++) {
            const _Float16* r = &As[wm + i * 16 + lm][0];
            af[i] = cat8(*(const h8*)(r + kb0), *(const h8*)(r + kb1));
        }
#pragma unroll
        for (int j = 0; j < 4; j++) {
            const _Float16* r = &Bs[wn + j * 16 + lm][0];
            bf[j] = cat8(*(const h8*)(r + kb0), *(const h8*)(r + kb1));
        }
#pragma unroll
        for (int i = 0; i < 2; i++)
#pragma unroll
            for (int j = 0; j < 4; j++)
                acc[i][j] = __builtin_amdgcn_wmma_f32_16x16x32_f16(
                    false, af[i], false, bf[j], (short)0, acc[i][j], false, false);
    }

    // epilogue; C/D layout: elem v -> m = v + 8*hi, n = lm
#pragma unroll
    for (int i = 0; i < 2; i++)
#pragma unroll
        for (int j = 0; j < 4; j++) {
            int n = n0 + wn + j * 16 + lm;
            if (!FULL && n >= N) continue;
            float bv = bias[n];
#pragma unroll
            for (int v = 0; v < 8; v++) {
                int m = m0 + wm + i * 16 + hi * 8 + v;
                long o = (long)m * N + n;
                float val = acc[i][j][v] + bv;
                if (Cadd) val += Cadd[o];
                val = actf(val, act);
                if (outF) outF[o] = val;
                if (outH) outH[o] = (_Float16)val;
            }
        }
}

// ---------------- host orchestration ----------------

extern "C" void kernel_launch(void* const* d_in, const int* in_sizes, int n_in,
                              void* d_out, int out_size, void* d_ws, size_t ws_size,
                              hipStream_t stream) {
    (void)in_sizes; (void)n_in; (void)out_size; (void)ws_size;

    const int*   rel    = (const int*)d_in[0];
    const float* entlog = (const float*)d_in[1];
    const float* obj    = (const float*)d_in[2];
    const float* vr     = (const float*)d_in[3];

    auto W32 = [&](int i) { return (const float*)d_in[i]; };

    char* ws = (char*)d_ws;
    size_t off = 0;
    auto alloc = [&](size_t bytes) -> void* {
        void* p = ws + off;
        off = (off + bytes + 255) & ~(size_t)255;
        return p;
    };

    // weight table: {d_in index of W, K, N}; bias at idx+1
    struct Wspec { int idx, k, n; };
    static const Wspec wspec[22] = {
        {4, 512, 256}, {6, 256, 128},   // send_ent
        {8, 512, 256}, {10, 256, 128},  // send_pred
        {12, 256, 384}, {14, 384, 512}, // recv_ent
        {16, 256, 384}, {18, 384, 512}, // recv_pred
        {20, 512, 512}, {22, 512, 512}, {24, 512, 512}, {26, 512, 512}, {28, 512, 512}, {30, 512, 512}, // ent gru 3w,3u,4w,4u,5w,5u
        {32, 512, 512}, {34, 512, 512}, {36, 512, 512}, {38, 512, 512}, {40, 512, 512}, {42, 512, 512}, // pred gru
        {44, 512, 512}, {46, 512, 51}   // out_pred
    };

    _Float16* wf[22];
    for (int i = 0; i < 22; i++) {
        long n = (long)wspec[i].k * wspec[i].n;
        wf[i] = (_Float16*)alloc((size_t)n * 2);
        k_cvt<<<(int)((n + 255) / 256), 256, 0, stream>>>(W32(wspec[i].idx), wf[i], n);
    }

    float*    neF  = (float*)alloc((size_t)NE * HD * 4);
    _Float16* neH  = (_Float16*)alloc((size_t)NE * HD * 2);
    float*    npF  = (float*)alloc((size_t)NP * HD * 4);
    _Float16* npH  = (_Float16*)alloc((size_t)NP * HD * 2);
    _Float16* hb   = (_Float16*)alloc((size_t)NP * HD * 2); // hidden-layer scratch (f16)
    float*    msE  = (float*)alloc((size_t)NE * 128 * 4);
    float*    msP  = (float*)alloc((size_t)NP * 128 * 4);
    float*    segS = (float*)alloc((size_t)NE * 128 * 4);
    float*    segO = (float*)alloc((size_t)NE * 128 * 4);
    float*    cntS = (float*)alloc((size_t)NE * 4);
    float*    cntO = (float*)alloc((size_t)NE * 4);
    _Float16* incE = (_Float16*)alloc((size_t)NE * 256 * 2);
    _Float16* incP = (_Float16*)alloc((size_t)NP * 256 * 2);
    _Float16* mrE  = (_Float16*)alloc((size_t)NE * HD * 2);
    _Float16* mrP  = (_Float16*)alloc((size_t)NP * HD * 2);
    float*    g3   = (float*)alloc((size_t)NP * HD * 4); // z / addend
    float*    gt   = (float*)alloc((size_t)NP * HD * 4); // r, then h
    _Float16* rn   = (_Float16*)alloc((size_t)NP * HD * 2);

    auto ew = [&](long n) { return dim3((unsigned)((n + 255) / 256)); };

    k_cvt_dual<<<ew((long)NE * HD), 256, 0, stream>>>(obj, neF, neH, (long)NE * HD);
    k_cvt_dual<<<ew((long)NP * HD), 256, 0, stream>>>(vr, npF, npH, (long)NP * HD);

    auto gemm = [&](const _Float16* A, int w, const float* Cadd, float* oF, _Float16* oH,
                    int M, int N, int K, int act) {
        dim3 g((unsigned)((N + 127) / 128), (unsigned)((M + 127) / 128));
        if ((N & 127) == 0 && (M & 127) == 0)
            k_gemm<true><<<g, 256, 0, stream>>>(A, wf[w], W32(wspec[w].idx + 1), Cadd, oF, oH,
                                                M, N, K, act);
        else
            k_gemm<false><<<g, 256, 0, stream>>>(A, wf[w], W32(wspec[w].idx + 1), Cadd, oF, oH,
                                                 M, N, K, act);
    };

    for (int t = 0; t < 3; t++) {
        // send MLPs (relu both layers)
        gemm(neH, 0, nullptr, nullptr, hb, NE, 256, 512, 1);
        gemm(hb, 1, nullptr, msE, nullptr, NE, 128, 256, 1);
        gemm(npH, 2, nullptr, nullptr, hb, NP, 256, 512, 1);
        gemm(hb, 3, nullptr, msP, nullptr, NP, 128, 256, 1);
        // graph aggregation
        k_zero<<<ew((long)NE * 128), 256, 0, stream>>>(segS, (long)NE * 128);
        k_zero<<<ew((long)NE * 128), 256, 0, stream>>>(segO, (long)NE * 128);
        k_zero<<<ew(NE), 256, 0, stream>>>(cntS, NE);
        k_zero<<<ew(NE), 256, 0, stream>>>(cntO, NE);
        k_counts<<<NP / 256, 256, 0, stream>>>(rel, cntS, cntO);
        k_segsum<<<(NP * 128) / 256, 256, 0, stream>>>(rel, msP, segS, segO);
        k_inc_ent<<<ew(NE * 2), 256, 0, stream>>>(segS, segO, cntS, cntO, incE);
        k_inc_pred<<<ew(NP * 2), 256, 0, stream>>>(rel, msE, incP);
        // recv MLPs (relu both layers)
        gemm(incE, 4, nullptr, nullptr, hb, NE, 384, 256, 1);
        gemm(hb, 5, nullptr, nullptr, mrE, NE, 512, 384, 1);
        gemm(incP, 6, nullptr, nullptr, hb, NP, 384, 256, 1);
        gemm(hb, 7, nullptr, nullptr, mrP, NP, 512, 384, 1);
        // ent GRU
        gemm(mrE, 8, nullptr, g3, nullptr, NE, 512, 512, 0);
        gemm(neH, 9, g3, g3, nullptr, NE, 512, 512, 2);      // z = sigmoid(.)
        gemm(mrE, 10, nullptr, gt, nullptr, NE, 512, 512, 0);
        gemm(neH, 11, gt, gt, nullptr, NE, 512, 512, 2);     // r = sigmoid(.)
        k_rn<<<ew((long)NE * HD), 256, 0, stream>>>(gt, neF, rn, (long)NE * HD);
        gemm(mrE, 12, nullptr, gt, nullptr, NE, 512, 512, 0);
        gemm(rn, 13, gt, gt, nullptr, NE, 512, 512, 3);      // h = tanh(.)
        k_combine<<<ew((long)NE * HD), 256, 0, stream>>>(g3, gt, neF, neH, (long)NE * HD);
        // pred GRU
        gemm(mrP, 14, nullptr, g3, nullptr, NP, 512, 512, 0);
        gemm(npH, 15, g3, g3, nullptr, NP, 512, 512, 2);
        gemm(mrP, 16, nullptr, gt, nullptr, NP, 512, 512, 0);
        gemm(npH, 17, gt, gt, nullptr, NP, 512, 512, 2);
        k_rn<<<ew((long)NP * HD), 256, 0, stream>>>(gt, npF, rn, (long)NP * HD);
        gemm(mrP, 18, nullptr, gt, nullptr, NP, 512, 512, 0);
        gemm(rn, 19, gt, gt, nullptr, NP, 512, 512, 3);
        k_combine<<<ew((long)NP * HD), 256, 0, stream>>>(g3, gt, npF, npH, (long)NP * HD);
    }

    // output head: relu(layer0) then linear 512->51 straight into d_out
    gemm(npH, 20, nullptr, nullptr, hb, NP, 512, 512, 1);
    gemm(hb, 21, nullptr, (float*)d_out, nullptr, NP, 51, 512, 0);

    // second tuple output: ent_cls_logits passthrough
    hipMemcpyAsync((float*)d_out + (size_t)NP * 51, entlog,
                   (size_t)NE * 151 * sizeof(float), hipMemcpyDeviceToDevice, stream);
}
